// LearnableGraphHead_3573412790296
// MI455X (gfx1250) — matrix operand order repaired and checked by reading, not compiled, not run on hardware
//
#include <hip/hip_runtime.h>

typedef __bf16 bf16;
typedef bf16 v16bf __attribute__((ext_vector_type(16)));
typedef float v8f __attribute__((ext_vector_type(8)));

#define B_ 16
#define T_ 512
#define DIN 1024
#define H_ 256
#define NN 64
#define M_ (B_ * T_)   // 8192
#define MN (B_ * NN)   // 1024
#define NCOEF 8

// ---------------- helpers ----------------
__device__ __forceinline__ bf16 f2bf(float f) {
  union { float f; unsigned u; } v; v.f = f;
  unsigned r = v.u + 0x7FFFu + ((v.u >> 16) & 1u);
  union { unsigned short s; bf16 b; } o; o.s = (unsigned short)(r >> 16);
  return o.b;
}
__device__ __forceinline__ float siluf(float x) { return x / (1.0f + expf(-x)); }
__device__ __forceinline__ float seluf(float x) {
  const float sc = 1.0507009873554805f, al = 1.6732632423543772f;
  return sc * (x > 0.0f ? x : al * (expf(x) - 1.0f));
}
__device__ __forceinline__ float eluf(float x) { return x > 0.0f ? x : expf(x) - 1.0f; }

// cubic B-spline bases, grid_range [-1,1], GRID_SIZE=5 -> 8 bases
__device__ __forceinline__ void bspline8(float x, float* outb) {
  const float h = 0.4f;
  float grid[12];
#pragma unroll
  for (int i = 0; i < 12; ++i) grid[i] = (float)(i - 3) * h - 1.0f;
  float b[11];
#pragma unroll
  for (int i = 0; i < 11; ++i) b[i] = (x >= grid[i] && x < grid[i + 1]) ? 1.0f : 0.0f;
#pragma unroll
  for (int k = 1; k <= 3; ++k) {
    float inv = 1.0f / ((float)k * h);
    for (int i = 0; i < 11 - k; ++i)
      b[i] = (x - grid[i]) * inv * b[i] + (grid[i + k + 1] - x) * inv * b[i + 1];
  }
#pragma unroll
  for (int i = 0; i < 8; ++i) outb[i] = b[i];
}

// ---------------- WMMA GEMM: C[M,N] (f32) = A[M,K] x B[K,N] (bf16 row-major) ----------------
#define BM 128
#define BN 128
#define BK 32
#define LDT 40  // padded LDS stride (elements); 80B, 16B aligned

__global__ __launch_bounds__(256) void gemm_bf16_wmma(
    const bf16* __restrict__ A, const bf16* __restrict__ B, float* __restrict__ C,
    int M, int N, int K, int accflag) {
  __shared__ __align__(16) bf16 Als[BM][LDT];  // [m][k]
  __shared__ __align__(16) bf16 Bls[BN][LDT];  // [n][k]  (transposed tile)
  const int tid = threadIdx.x;
  const int bm = blockIdx.x * BM;
  const int bn = blockIdx.y * BN;
  const int wave = tid >> 5, lane = tid & 31;
  const int wm = wave >> 2, wn = wave & 3;      // 2 x 4 wave grid
  const int lrow = lane & 15, lhi = lane >> 4;

  v8f acc[4][2];
#pragma unroll
  for (int i = 0; i < 4; ++i)
#pragma unroll
    for (int j = 0; j < 2; ++j)
#pragma unroll
      for (int r = 0; r < 8; ++r) acc[i][j][r] = 0.0f;

  const int ar = tid & 127;           // A row in tile
  const int ac = (tid >> 7) * 16;     // A col base (0/16)
  const int bk2 = (tid & 15) * 2;     // B k-row pair base (0..30)
  const int bnb8 = (tid >> 4) * 8;    // B n base (0..120)

  const int nt = K / BK;

  // preload tile 0 into registers
  uint4 a0, a1, b0, b1;
  {
    const bf16* ga = A + (size_t)(bm + ar) * K + ac;
    a0 = *(const uint4*)(ga);
    a1 = *(const uint4*)(ga + 8);
    const bf16* gb = B + (size_t)bk2 * N + bn + bnb8;
    b0 = *(const uint4*)(gb);
    b1 = *(const uint4*)(gb + N);
  }

  for (int it = 0; it < nt; ++it) {
    // ---- stage registers -> LDS ----
    *(uint4*)(&Als[ar][ac])     = a0;
    *(uint4*)(&Als[ar][ac + 8]) = a1;
    {
      const unsigned short* s0 = (const unsigned short*)&b0;
      const unsigned short* s1 = (const unsigned short*)&b1;
#pragma unroll
      for (int j = 0; j < 8; ++j) {
        unsigned v = (unsigned)s0[j] | ((unsigned)s1[j] << 16);
        *(unsigned*)(&Bls[bnb8 + j][bk2]) = v;  // packs (k, k+1) for column n
      }
    }
    __syncthreads();

    // ---- prefetch next tile into registers (overlaps with WMMA below) ----
    if (it + 1 < nt) {
      const bf16* gan = A + (size_t)(bm + ar) * K + (it + 1) * BK + ac;
      a0 = *(const uint4*)(gan);
      a1 = *(const uint4*)(gan + 8);
      const bf16* gbn = B + (size_t)((it + 1) * BK + bk2) * N + bn + bnb8;
      b0 = *(const uint4*)(gbn);
      b1 = *(const uint4*)(gbn + N);
      if (it + 2 < nt) {
        __builtin_prefetch(A + (size_t)(bm + ar) * K + (it + 2) * BK + ac, 0, 1);
        __builtin_prefetch(B + (size_t)((it + 2) * BK + bk2) * N + bn + bnb8, 0, 1);
      }
    }

    // ---- fragments + WMMA ----
    v16bf bfrag[2];
#pragma unroll
    for (int sn = 0; sn < 2; ++sn) {
      const bf16* p = &Bls[wn * 32 + sn * 16 + lrow][lhi * 8];
#pragma unroll
      for (int j = 0; j < 8; ++j) { bfrag[sn][j] = p[j]; bfrag[sn][8 + j] = p[16 + j]; }
    }
#pragma unroll
    for (int sm = 0; sm < 4; ++sm) {
      const bf16* pa = &Als[wm * 64 + sm * 16 + lrow][lhi * 8];
      v16bf afrag;
#pragma unroll
      for (int j = 0; j < 8; ++j) { afrag[j] = pa[j]; afrag[8 + j] = pa[16 + j]; }
#pragma unroll
      for (int sn = 0; sn < 2; ++sn) {
        acc[sm][sn] = __builtin_amdgcn_wmma_f32_16x16x32_bf16(
            false, afrag, false, bfrag[sn], (short)0, acc[sm][sn], false, false);
      }
    }
    __syncthreads();
  }

  // ---- epilogue: accumulate decision hoisted to a single uniform branch ----
  if (accflag) {
#pragma unroll
    for (int sm = 0; sm < 4; ++sm)
#pragma unroll
      for (int sn = 0; sn < 2; ++sn) {
        int col = bn + wn * 32 + sn * 16 + lrow;
        int rbase = bm + wm * 64 + sm * 16 + lhi * 8;
#pragma unroll
        for (int r = 0; r < 8; ++r) {
          size_t off = (size_t)(rbase + r) * N + col;
          C[off] = acc[sm][sn][r] + C[off];
        }
      }
  } else {
#pragma unroll
    for (int sm = 0; sm < 4; ++sm)
#pragma unroll
      for (int sn = 0; sn < 2; ++sn) {
        int col = bn + wn * 32 + sn * 16 + lrow;
        int rbase = bm + wm * 64 + sm * 16 + lhi * 8;
#pragma unroll
        for (int r = 0; r < 8; ++r) {
          size_t off = (size_t)(rbase + r) * N + col;
          C[off] = acc[sm][sn][r];
        }
      }
  }
}

// ---------------- prep kernels ----------------
// Wa[k = c*1024 + i][o], c==0: base_w; c>=1: spline_w*scaler
__global__ void k_prep_wa(const float* __restrict__ bw, const float* __restrict__ sw,
                          const float* __restrict__ sc, bf16* __restrict__ Wa) {
  int idx = blockIdx.x * 256 + threadIdx.x;  // 9216*256
  int k = idx >> 8, o = idx & 255;
  int c = k >> 10, i = k & 1023;
  float v;
  if (c == 0) v = bw[o * DIN + i];
  else v = sw[((size_t)(o * DIN + i)) * NCOEF + (c - 1)] * sc[o * DIN + i];
  Wa[idx] = f2bf(v);
}

// Wc[k = dt*256 + i][o] = conv_w[o, i, dt]
__global__ void k_prep_wconv(const float* __restrict__ cw, bf16* __restrict__ Wc) {
  int idx = blockIdx.x * 256 + threadIdx.x;  // 768*256
  int k = idx >> 8, o = idx & 255;
  int dt = k >> 8, i = k & 255;
  Wc[idx] = f2bf(cw[(o * H_ + i) * 3 + dt]);
}

// augmented activation plane c: c==0 -> silu(x); c>=1 -> spline basis (c-1)
__global__ void k_prep_plane(const float* __restrict__ x, bf16* __restrict__ Ap, int c) {
  int idx = blockIdx.x * 256 + threadIdx.x;  // 8192*1024
  float f = x[idx];
  float v;
  if (c == 0) v = siluf(f);
  else { float bb[8]; bspline8(f, bb); v = bb[c - 1]; }
  Ap[idx] = f2bf(v);
}

// im2col for conv1d(k=3, pad=1): src [B,T,H] f32 -> dst [B*T, 3*H] bf16
__global__ void k_im2col(const float* __restrict__ src, bf16* __restrict__ dst) {
  int idx = blockIdx.x * 256 + threadIdx.x;  // 8192*768
  int m = idx / 768, k = idx - m * 768;
  int dt = k >> 8, i = k & 255;
  int b = m >> 9, t = m & 511;
  int tt = t + dt - 1;
  float v = (tt >= 0 && tt < T_) ? src[((size_t)(b * T_ + tt)) * H_ + i] : 0.0f;
  dst[idx] = f2bf(v);
}

__global__ void k_selu_bias(float* __restrict__ dat, const float* __restrict__ bias) {
  int idx = blockIdx.x * 256 + threadIdx.x;  // 8192*256
  int o = idx & 255;
  dat[idx] = seluf(dat[idx] + bias[o]);
}

// adaptive avg pool (8 taps) + pos_emb : y2 [B,T,H] -> nodes [B,64,H]
__global__ void k_pool_pos(const float* __restrict__ src, const float* __restrict__ pos,
                           float* __restrict__ nodes) {
  int idx = blockIdx.x * 256 + threadIdx.x;  // 16*64*256
  int o = idx & 255, n = (idx >> 8) & 63, b = idx >> 14;
  size_t base = ((size_t)(b * T_ + n * 8)) * H_ + o;
  float s = 0.0f;
#pragma unroll
  for (int r = 0; r < 8; ++r) s += src[base + (size_t)r * H_];
  nodes[idx] = s * 0.125f + pos[n * H_ + o];
}

__global__ void k_cast_bf16(const float* __restrict__ src, bf16* __restrict__ dst, int n) {
  int idx = blockIdx.x * 256 + threadIdx.x;
  if (idx < n) dst[idx] = f2bf(src[idx]);
}

// scores[b,n,m] = <nodes[b,n,:], nodes[b,m,:]>
__global__ void k_scores(const float* __restrict__ nodes, float* __restrict__ scores) {
  int row = blockIdx.x;  // b*64+n
  int tid = threadIdx.x;
  int b = row >> 6;
  __shared__ float hv[256];
  __shared__ float part[256];
  hv[tid] = nodes[(size_t)row * H_ + tid];
  __syncthreads();
  int m = tid & 63, p = tid >> 6;
  const float* hm = nodes + (size_t)(b * NN + m) * H_;
  float s = 0.0f;
  for (int d = p * 64; d < p * 64 + 64; ++d) s += hv[d] * hm[d];
  part[tid] = s;
  __syncthreads();
  if (tid < 64)
    scores[(size_t)row * NN + m] = part[m] + part[m + 64] + part[m + 128] + part[m + 192];
}

// top-8 per row -> one-hot rows
__global__ void k_topk(const float* __restrict__ scores, float* __restrict__ adjA) {
  int row = blockIdx.x;
  int tid = threadIdx.x;  // 64 threads
  __shared__ float s[64];
  s[tid] = scores[(size_t)row * NN + tid];
  adjA[(size_t)row * NN + tid] = 0.0f;
  __syncthreads();
  if (tid == 0) {
    for (int it = 0; it < 8; ++it) {
      int best = 0; float bv = s[0];
      for (int j = 1; j < 64; ++j) if (s[j] > bv) { bv = s[j]; best = j; }
      adjA[(size_t)row * NN + best] = 1.0f;
      s[best] = -3.0e38f;
    }
  }
}

// adj = max(A, A^T); diag = 1
__global__ void k_sym(const float* __restrict__ adjA, float* __restrict__ adj) {
  int row = blockIdx.x;
  int m = threadIdx.x;  // 64 threads
  int b = row >> 6, n = row & 63;
  float v = fmaxf(adjA[(size_t)row * NN + m], adjA[(size_t)(b * NN + m) * NN + n]);
  if (m == n) v = 1.0f;
  adj[(size_t)row * NN + m] = v;
}

// per-node attention coefficients: asrc = <hW, a_src>, adst = <hW, a_dst>
__global__ void k_rowdot(const float* __restrict__ hW, const float* __restrict__ asr,
                         const float* __restrict__ ads, float* __restrict__ asrc,
                         float* __restrict__ adst) {
  int row = blockIdx.x, tid = threadIdx.x;
  __shared__ float p1[256], p2[256];
  float hv = hW[(size_t)row * H_ + tid];
  p1[tid] = hv * asr[tid];
  p2[tid] = hv * ads[tid];
  __syncthreads();
  for (int s = 128; s > 0; s >>= 1) {
    if (tid < s) { p1[tid] += p1[tid + s]; p2[tid] += p2[tid + s]; }
    __syncthreads();
  }
  if (tid == 0) { asrc[row] = p1[0]; adst[row] = p2[0]; }
}

// masked leaky-relu + softmax over m
__global__ void k_softmax_e(const float* __restrict__ asrc, const float* __restrict__ adst,
                            const float* __restrict__ adj, float* __restrict__ attn) {
  int row = blockIdx.x;
  int m = threadIdx.x;  // 64 threads
  int b = row >> 6;
  __shared__ float red[64];
  float e = asrc[row] + adst[b * NN + m];
  e = e > 0.0f ? e : 0.2f * e;
  if (!(adj[(size_t)row * NN + m] > 0.0f)) e = -1.0e9f;
  red[m] = e;
  __syncthreads();
  for (int s = 32; s > 0; s >>= 1) {
    if (m < s) red[m] = fmaxf(red[m], red[m + s]);
    __syncthreads();
  }
  float emax = red[0];
  __syncthreads();
  float ex = expf(e - emax);
  red[m] = ex;
  __syncthreads();
  for (int s = 32; s > 0; s >>= 1) {
    if (m < s) red[m] += red[m + s];
    __syncthreads();
  }
  attn[(size_t)row * NN + m] = ex / red[0];
}

// nodes += elu(attn @ hW)  (residual)
__global__ void k_aggregate(const float* __restrict__ attn, const float* __restrict__ hW,
                            float* __restrict__ nodes) {
  int row = blockIdx.x, tid = threadIdx.x;
  int b = row >> 6;
  __shared__ float at[64];
  if (tid < 64) at[tid] = attn[(size_t)row * NN + tid];
  __syncthreads();
  float acc = 0.0f;
  for (int m = 0; m < 64; ++m) acc += at[m] * hW[(size_t)(b * NN + m) * H_ + tid];
  nodes[(size_t)row * H_ + tid] += eluf(acc);
}

__global__ void k_layernorm(float* __restrict__ nodes, const float* __restrict__ g,
                            const float* __restrict__ bb) {
  int row = blockIdx.x, tid = threadIdx.x;
  __shared__ float red[256];
  float v = nodes[(size_t)row * H_ + tid];
  red[tid] = v;
  __syncthreads();
  for (int s = 128; s > 0; s >>= 1) { if (tid < s) red[tid] += red[tid + s]; __syncthreads(); }
  float mu = red[0] * (1.0f / 256.0f);
  __syncthreads();
  float d = v - mu;
  red[tid] = d * d;
  __syncthreads();
  for (int s = 128; s > 0; s >>= 1) { if (tid < s) red[tid] += red[tid + s]; __syncthreads(); }
  float var = red[0] * (1.0f / 256.0f);
  nodes[(size_t)row * H_ + tid] = d * rsqrtf(var + 1e-5f) * g[tid] + bb[tid];
}

// emb[b, 0:256]=mean_n, emb[b, 256:512]=max_n -> out+32
__global__ void k_emb(const float* __restrict__ nodes, float* __restrict__ out) {
  int b = blockIdx.x, o = threadIdx.x;
  float sm = 0.0f, mx = -3.0e38f;
  for (int n = 0; n < NN; ++n) {
    float v = nodes[(size_t)(b * NN + n) * H_ + o];
    sm += v;
    mx = fmaxf(mx, v);
  }
  out[32 + b * 512 + o] = sm * (1.0f / 64.0f);
  out[32 + b * 512 + 256 + o] = mx;
}

// classifier KAN: logits[b,o], emb at out+32
__global__ void k_classifier(const float* __restrict__ emb, const float* __restrict__ cbw,
                             const float* __restrict__ csw, const float* __restrict__ csc,
                             float* __restrict__ out) {
  int bo = blockIdx.x;  // 0..31
  int b = bo >> 1, o = bo & 1;
  int tid = threadIdx.x;
  __shared__ float red[256];
  float acc = 0.0f;
  for (int j = tid; j < 512; j += 256) {
    float e = emb[b * 512 + j];
    float s = siluf(e) * cbw[o * 512 + j];
    float bbv[8];
    bspline8(e, bbv);
    float sc = csc[o * 512 + j];
#pragma unroll
    for (int c = 0; c < 8; ++c) s += bbv[c] * csw[(size_t)(o * 512 + j) * 8 + c] * sc;
    acc += s;
  }
  red[tid] = acc;
  __syncthreads();
  for (int s = 128; s > 0; s >>= 1) { if (tid < s) red[tid] += red[tid + s]; __syncthreads(); }
  if (tid == 0) out[b * 2 + o] = red[0];
}

// ---------------- launcher ----------------
extern "C" void kernel_launch(void* const* d_in, const int* in_sizes, int n_in,
                              void* d_out, int out_size, void* d_ws, size_t ws_size,
                              hipStream_t stream) {
  (void)in_sizes; (void)n_in; (void)out_size; (void)ws_size;
  const float* x   = (const float*)d_in[0];
  const float* bbw = (const float*)d_in[1];
  const float* bsw = (const float*)d_in[2];
  const float* bsc = (const float*)d_in[3];
  const float* c1w = (const float*)d_in[4];
  const float* c1b = (const float*)d_in[5];
  const float* c2w = (const float*)d_in[6];
  const float* c2b = (const float*)d_in[7];
  const float* pos = (const float*)d_in[8];
  const float* gW  = (const float*)d_in[9];
  const float* gas = (const float*)d_in[10];
  const float* gad = (const float*)d_in[11];
  const float* lng = (const float*)d_in[12];
  const float* lnb = (const float*)d_in[13];
  const float* cbw = (const float*)d_in[14];
  const float* csw = (const float*)d_in[15];
  const float* csc = (const float*)d_in[16];
  float* out = (float*)d_out;

  char* wptr = (char*)d_ws;
  auto alloc = [&](size_t bytes) -> void* {
    void* p = (void*)wptr;
    wptr += (bytes + 255) & ~(size_t)255;
    return p;
  };
  bf16* Wa     = (bf16*)alloc((size_t)9 * DIN * H_ * 2);
  bf16* Wc1    = (bf16*)alloc((size_t)768 * H_ * 2);
  bf16* Wc2    = (bf16*)alloc((size_t)768 * H_ * 2);
  bf16* Apl    = (bf16*)alloc((size_t)M_ * DIN * 2);
  bf16* Xcol   = (bf16*)alloc((size_t)M_ * 768 * 2);
  float* h     = (float*)alloc((size_t)M_ * H_ * 4);
  float* y1    = (float*)alloc((size_t)M_ * H_ * 4);
  float* nodes = (float*)alloc((size_t)MN * H_ * 4);
  bf16* nodesB = (bf16*)alloc((size_t)MN * H_ * 2);
  bf16* WB     = (bf16*)alloc((size_t)H_ * H_ * 2);
  float* hW    = (float*)alloc((size_t)MN * H_ * 4);
  float* scr   = (float*)alloc((size_t)B_ * NN * NN * 4);
  float* adjA  = (float*)alloc((size_t)B_ * NN * NN * 4);
  float* adj   = (float*)alloc((size_t)B_ * NN * NN * 4);
  float* asrc  = (float*)alloc((size_t)MN * 4);
  float* adst  = (float*)alloc((size_t)MN * 4);
  float* attn  = (float*)alloc((size_t)B_ * NN * NN * 4);

  // weight packing (bf16)
  k_prep_wa<<<9 * DIN, 256, 0, stream>>>(bbw, bsw, bsc, Wa);
  k_prep_wconv<<<768, 256, 0, stream>>>(c1w, Wc1);
  k_prep_wconv<<<768, 256, 0, stream>>>(c2w, Wc2);

  // bridge KAN: 9 accumulating WMMA GEMMs (silu + 8 spline planes)
  for (int c = 0; c < 9; ++c) {
    k_prep_plane<<<M_ * DIN / 256, 256, 0, stream>>>(x, Apl, c);
    gemm_bf16_wmma<<<dim3(M_ / BM, H_ / BN), 256, 0, stream>>>(
        Apl, Wa + (size_t)c * DIN * H_, h, M_, H_, DIN, c > 0 ? 1 : 0);
  }

  // conv1 -> selu -> conv2 -> selu (im2col WMMA GEMMs)
  k_im2col<<<M_ * 768 / 256, 256, 0, stream>>>(h, Xcol);
  gemm_bf16_wmma<<<dim3(M_ / BM, H_ / BN), 256, 0, stream>>>(Xcol, Wc1, y1, M_, H_, 768, 0);
  k_selu_bias<<<M_, 256, 0, stream>>>(y1, c1b);
  k_im2col<<<M_ * 768 / 256, 256, 0, stream>>>(y1, Xcol);
  gemm_bf16_wmma<<<dim3(M_ / BM, H_ / BN), 256, 0, stream>>>(Xcol, Wc2, h, M_, H_, 768, 0);
  k_selu_bias<<<M_, 256, 0, stream>>>(h, c2b);

  // pool + pos, adjacency
  k_pool_pos<<<MN, 256, 0, stream>>>(h, pos, nodes);
  k_scores<<<MN, 256, 0, stream>>>(nodes, scr);
  k_topk<<<MN, 64, 0, stream>>>(scr, adjA);
  k_sym<<<MN, 64, 0, stream>>>(adjA, adj);

  // residual GAT layers
  for (int l = 0; l < 2; ++l) {
    k_cast_bf16<<<MN * H_ / 256, 256, 0, stream>>>(nodes, nodesB, MN * H_);
    k_cast_bf16<<<H_ * H_ / 256, 256, 0, stream>>>(gW + (size_t)l * H_ * H_, WB, H_ * H_);
    gemm_bf16_wmma<<<dim3(MN / BM, H_ / BN), 256, 0, stream>>>(nodesB, WB, hW, MN, H_, H_, 0);
    k_rowdot<<<MN, 256, 0, stream>>>(hW, gas + l * H_, gad + l * H_, asrc, adst);
    k_softmax_e<<<MN, 64, 0, stream>>>(asrc, adst, adj, attn);
    k_aggregate<<<MN, 256, 0, stream>>>(attn, hW, nodes);
  }

  // layernorm, embedding, classifier
  k_layernorm<<<MN, 256, 0, stream>>>(nodes, lng, lnb);
  k_emb<<<B_, 256, 0, stream>>>(nodes, out);
  k_classifier<<<B_ * 2, 256, 0, stream>>>(out + 32, cbw, csw, csc, out);
}
